// TopKSparseAutoencoder_10874857194038
// MI455X (gfx1250) — matrix operand order repaired and checked by reading, not compiled, not run on hardware
//
#include <hip/hip_runtime.h>
#include <hip/hip_bf16.h>

// TopK Sparse Autoencoder forward, fused for MI455X (gfx1250, wave32).
//   x:     [4096, 768]   f32
//   W_enc: [768, 24576]  f32
//   b_enc: [24576]       f32
//   W_dec: [24576, 768]  f32
//   b_dec: [768]         f32
// out = recon [4096*768] ++ acts [4096*24576] ++ topk_idx [4096*32] (as f32)
//
// Pipeline:
//   K1: swizzle W_enc f32 -> bf16 into d_ws in WMMA B-fragment order
//   K2: zero the 402 MB acts region (HBM floor ~17 us @ 23.3 TB/s)
//   K3: bf16-WMMA encoder GEMM (32 rows/WG; rows 0-15's A fragments pinned in
//       192 VGPRs, rows 16-31's read from LDS, sharing each B fragment).
//       Each row-tile is split across 2 WGs (half the latents each) -> 256 WGs
//       of grid parallelism at identical L2 traffic. Per-half top-32 to ws.
//   K4: merge the 2x32 candidates/row -> global top-32 (sorted desc, ties by
//       lower index, matching jax.lax.top_k), scatter acts, emit indices,
//       sparse decode recon.

#define D_INP   768
#define NLAT    24576
#define TOPK    32
#define BATCH   4096

#define TILE_M  32          // rows per workgroup (2 x 16-row halves)
#define TILE_N  128         // latent columns per chunk (8 waves x 16)
#define KSTEP   32          // bf16 WMMA K
#define KB      (D_INP / KSTEP)   // 24 k-blocks
#define THREADS 256         // 8 wave32s
#define A_STRIDE (D_INP + 8)
#define NSPLIT  2           // latent-dimension split (WGs per row tile)
#define NCHUNK_HALF (NLAT / TILE_N / NSPLIT)   // 96 chunks per half

typedef __attribute__((ext_vector_type(16))) __bf16 v16bf;
typedef __attribute__((ext_vector_type(8)))  float  v8f;

union Frag16 {
    v16bf v;
    uint4 q[2];
    unsigned int u[8];
    unsigned short s[16];
};

// f32 -> bf16 (round to nearest even)
__device__ __forceinline__ unsigned short f2bf(float f) {
    unsigned int u = __builtin_bit_cast(unsigned int, f);
    unsigned int lsb = (u >> 16) & 1u;
    u += 0x7fffu + lsb;
    return (unsigned short)(u >> 16);
}

// ---------------------------------------------------------------------------
// K1: convert W_enc [768, 24576] f32 -> bf16 in d_ws, pre-swizzled into the
// contiguous-k (k = khalf*16 + j) wave32 B-fragment layout so the GEMM loads fragments as b128s:
//   ws[ ((nb*KB + kb)*2 + khalf)*256 + l15*16 + j ]
//     = bf16( W_enc[ kb*32 + khalf*16 + j ][ nb*16 + l15 ] )
// ---------------------------------------------------------------------------
__global__ void __launch_bounds__(THREADS)
sae_swizzle_wenc(const float* __restrict__ Wenc, unsigned short* __restrict__ wsB) {
    __shared__ unsigned short tile[32][130];   // [k within block][n within 128]
    const int tid = threadIdx.x;
    const int nbg = blockIdx.x;                // 128-column group
    const int kb  = blockIdx.y;                // 32-row k block

    for (int i = tid; i < 32 * 128; i += THREADS) {
        int kk = i >> 7, nn = i & 127;
        tile[kk][nn] = f2bf(Wenc[(size_t)(kb * 32 + kk) * NLAT + nbg * 128 + nn]);
    }
    __syncthreads();

    const int l15 = tid >> 4;                  // dst n index
    const int j   = tid & 15;                  // dst k index
    for (int w = 0; w < 16; ++w) {
        int nb2   = w >> 1;
        int khalf = w & 1;
        size_t dst = ((((size_t)(nbg * 8 + nb2)) * KB + kb) * 2 + khalf) * 256 + tid;
        wsB[dst] = tile[khalf * 16 + j][nb2 * 16 + l15];
    }
}

// ---------------------------------------------------------------------------
// K2: zero the acts output region with wide stores.
// ---------------------------------------------------------------------------
__global__ void __launch_bounds__(THREADS)
sae_zero_acts(float4* __restrict__ p, long n4) {
    long i = (long)blockIdx.x * blockDim.x + threadIdx.x;
    long stride = (long)gridDim.x * blockDim.x;
    float4 z; z.x = 0.f; z.y = 0.f; z.z = 0.f; z.w = 0.f;
    for (; i < n4; i += stride) p[i] = z;
}

// ---------------------------------------------------------------------------
// K3: encoder GEMM + per-half top-32 (32 rows / WG, half of latents / WG).
//   blockIdx.x = rowtile * NSPLIT + half
// ---------------------------------------------------------------------------
__global__ void __launch_bounds__(THREADS)
sae_encode_topk(const float* __restrict__ x, const unsigned short* __restrict__ wsB,
                const float* __restrict__ benc,
                float* __restrict__ wsTopV, int* __restrict__ wsTopI)
{
    __shared__ unsigned short As[16][A_STRIDE];       // bf16 x-tile (one half)
    __shared__ float          tkV[TILE_M][TOPK];
    __shared__ int            tkI[TILE_M][TOPK];
    __shared__ float          pV[TILE_M][TILE_N];     // pending candidates
    __shared__ unsigned short pI[TILE_M][TILE_N];     // column idx fits u16
    __shared__ float rMin[TILE_M];
    __shared__ int   rMinPos[TILE_M];
    __shared__ int   rCnt[TILE_M];
    __shared__ int   rPend[TILE_M];

    const int tid   = threadIdx.x;
    const int rt    = blockIdx.x / NSPLIT;            // row tile
    const int half  = blockIdx.x % NSPLIT;            // latent half
    const int row0  = rt * TILE_M;
    const int wave  = tid >> 5;
    const int lane  = tid & 31;
    const int l15   = lane & 15;
    const int khalf = lane >> 4;          // 0: lanes 0-15, 1: lanes 16-31

    if (tid < TILE_M) {
        rCnt[tid] = 0; rPend[tid] = 0;
        rMin[tid] = -3.4e38f; rMinPos[tid] = 0;
    }

    // --- Stage rows 0..15 of the x tile, preload their A fragments to regs ---
    for (int i = tid; i < 16 * D_INP; i += THREADS) {
        int r = i / D_INP, c = i - r * D_INP;
        As[r][c] = f2bf(x[(size_t)(row0 + r) * D_INP + c]);
    }
    __syncthreads();

    Frag16 aF[KB];                       // 24 x 8 = 192 VGPRs, rows 0..15
    #pragma unroll
    for (int kb = 0; kb < KB; ++kb) {
        const int k0 = kb * KSTEP;
        #pragma unroll
        for (int i2 = 0; i2 < 8; ++i2) {
            int kOff = k0 + (khalf << 4) + (i2 << 1);
            aF[kb].u[i2] = *(const unsigned int*)&As[l15][kOff];
        }
    }
    __syncthreads();

    // --- Overwrite LDS with rows 16..31 (read per k-step inside the loop) ---
    for (int i = tid; i < 16 * D_INP; i += THREADS) {
        int r = i / D_INP, c = i - r * D_INP;
        As[r][c] = f2bf(x[(size_t)(row0 + 16 + r) * D_INP + c]);
    }
    __syncthreads();

    for (int cidx = 0; cidx < NCHUNK_HALF; ++cidx) {
        const int nchunk = half * NCHUNK_HALF + cidx;
        const int nbG  = nchunk * 8 + wave;           // global 16-col block
        const int ncol = (nbG << 4) + l15;            // this lane's column
        const unsigned short* bp =
            wsB + (((size_t)nbG * KB) * 2 + khalf) * 256 + (l15 << 4);

        v8f acc0 = {};   // rows 0..15
        v8f acc1 = {};   // rows 16..31
        #pragma unroll
        for (int kb = 0; kb < KB; ++kb) {
            Frag16 b;
            b.q[0] = *(const uint4*)(bp + (size_t)kb * 512);
            b.q[1] = *(const uint4*)(bp + (size_t)kb * 512 + 8);

            Frag16 aH;                    // rows 16..31 fragment from LDS
            const int k0 = kb * KSTEP;
            #pragma unroll
            for (int i2 = 0; i2 < 8; ++i2) {
                int kOff = k0 + (khalf << 4) + (i2 << 1);
                aH.u[i2] = *(const unsigned int*)&As[l15][kOff];
            }

            acc0 = __builtin_amdgcn_wmma_f32_16x16x32_bf16(
                false, aF[kb].v, false, b.v, (short)0, acc0, false, false);
            acc1 = __builtin_amdgcn_wmma_f32_16x16x32_bf16(
                false, aH.v, false, b.v, (short)0, acc1, false, false);
        }

        // Threshold filter straight from accumulators:
        // lane holds column `ncol`; acc0[r] is row khalf*8+r, acc1[r] is +16.
        const float bias = benc[ncol];
        #pragma unroll
        for (int r = 0; r < 8; ++r) {
            {
                int   row = (khalf << 3) + r;
                float v   = acc0[r] + bias;
                float thr = (rCnt[row] < TOPK) ? -3.4e38f : rMin[row];
                if (v > thr) {
                    int p = atomicAdd(&rPend[row], 1);
                    pV[row][p] = v;
                    pI[row][p] = (unsigned short)ncol;
                }
            }
            {
                int   row = 16 + (khalf << 3) + r;
                float v   = acc1[r] + bias;
                float thr = (rCnt[row] < TOPK) ? -3.4e38f : rMin[row];
                if (v > thr) {
                    int p = atomicAdd(&rPend[row], 1);
                    pV[row][p] = v;
                    pI[row][p] = (unsigned short)ncol;
                }
            }
        }
        __syncthreads();

        // Serial merge of the (few) surviving candidates, one thread per row.
        // Expected survivors over a half-row: ~K*ln(N/(2K)) ~= 190 of 12288.
        if (tid < TILE_M) {
            int   row = tid;
            int   cnt = rCnt[row];
            float mn  = rMin[row];
            int   mp  = rMinPos[row];
            int   np  = rPend[row];
            for (int p = 0; p < np; ++p) {
                float v  = pV[row][p];
                int   ix = (int)pI[row][p];
                if (cnt < TOPK) {
                    tkV[row][cnt] = v; tkI[row][cnt] = ix; ++cnt;
                    if (cnt == TOPK) {
                        mn = tkV[row][0]; mp = 0;
                        for (int q = 1; q < TOPK; ++q)
                            if (tkV[row][q] < mn) { mn = tkV[row][q]; mp = q; }
                    }
                } else if (v > mn) {
                    tkV[row][mp] = v; tkI[row][mp] = ix;
                    mn = tkV[row][0]; mp = 0;
                    for (int q = 1; q < TOPK; ++q)
                        if (tkV[row][q] < mn) { mn = tkV[row][q]; mp = q; }
                }
            }
            rCnt[row] = cnt; rMin[row] = mn; rMinPos[row] = mp; rPend[row] = 0;
        }
        __syncthreads();
    }

    // Dump this half's per-row top-32 (unsorted) to workspace:
    //   wsTop[(grow*NSPLIT + half)*TOPK + e]
    {
        const int row  = tid >> 3;
        const int t8   = tid & 7;
        const int grow = row0 + row;
        for (int e = t8; e < TOPK; e += 8) {
            size_t o = ((size_t)grow * NSPLIT + half) * TOPK + e;
            wsTopV[o] = tkV[row][e];
            wsTopI[o] = tkI[row][e];
        }
    }
}

// ---------------------------------------------------------------------------
// K4: merge NSPLIT x 32 candidates per row -> global top-32 (sorted desc,
// ties broken by lower index to match jax.lax.top_k), then scatter acts,
// emit indices (as f32), and sparse-decode recon.
// ---------------------------------------------------------------------------
__global__ void __launch_bounds__(THREADS)
sae_merge_decode(const float* __restrict__ wsTopV, const int* __restrict__ wsTopI,
                 const float* __restrict__ Wdec, const float* __restrict__ bdec,
                 float* __restrict__ out)
{
    const int NC = NSPLIT * TOPK;                 // 64 candidates per row
    __shared__ float cV[TILE_M][NSPLIT * TOPK];
    __shared__ int   cI[TILE_M][NSPLIT * TOPK];

    const int tid  = threadIdx.x;
    const int row0 = blockIdx.x * TILE_M;

    // Cooperative load: halves are contiguous -> row-linear in ws.
    for (int i = tid; i < TILE_M * NC; i += THREADS) {
        int row = i / NC, e = i - row * NC;
        cV[row][e] = wsTopV[(size_t)(row0 + row) * NC + e];
        cI[row][e] = wsTopI[(size_t)(row0 + row) * NC + e];
    }
    __syncthreads();

    // Insertion-sort all 64 candidates descending; first 32 = global top-k.
    if (tid < TILE_M) {
        int row = tid;
        for (int i2 = 1; i2 < NC; ++i2) {
            float v  = cV[row][i2];
            int   ix = cI[row][i2];
            int j = i2 - 1;
            while (j >= 0 && (cV[row][j] < v ||
                              (cV[row][j] == v && cI[row][j] > ix))) {
                cV[row][j + 1] = cV[row][j];
                cI[row][j + 1] = cI[row][j];
                --j;
            }
            cV[row][j + 1] = v;
            cI[row][j + 1] = ix;
        }
    }
    __syncthreads();

    float* recon = out;
    float* acts  = out + (size_t)BATCH * D_INP;
    float* oidx  = out + (size_t)BATCH * D_INP + (size_t)BATCH * NLAT;

    const int row  = tid >> 3;          // 8 threads per row, 32 rows
    const int t8   = tid & 7;
    const int grow = row0 + row;

    // Scatter relu(topk_vals) into acts; emit topk indices (as f32).
    for (int e = t8; e < TOPK; e += 8) {
        float v  = cV[row][e];
        int   ix = cI[row][e];
        acts[(size_t)grow * NLAT + ix] = v > 0.f ? v : 0.f;
        oidx[(size_t)grow * TOPK + e]  = (float)ix;
    }

    // Sparse decode: recon[row] = b_dec + sum_e relu(v_e) * W_dec[idx_e, :]
    for (int c = t8; c < D_INP; c += 8) {
        float a = bdec[c];
        for (int e = 0; e < TOPK; ++e) {
            float v = cV[row][e];
            v = v > 0.f ? v : 0.f;
            a += v * Wdec[(size_t)cI[row][e] * D_INP + c];
        }
        recon[(size_t)grow * D_INP + c] = a;
    }
}

extern "C" void kernel_launch(void* const* d_in, const int* in_sizes, int n_in,
                              void* d_out, int out_size, void* d_ws, size_t ws_size,
                              hipStream_t stream) {
    const float* x    = (const float*)d_in[0];
    const float* Wenc = (const float*)d_in[1];
    const float* benc = (const float*)d_in[2];
    const float* Wdec = (const float*)d_in[3];
    const float* bdec = (const float*)d_in[4];
    float* out = (float*)d_out;

    // Workspace layout:
    //   [0, 37.75 MB)  : swizzled bf16 W_enc
    //   then           : per-half top-k values (4096*2*32 f32 = 1 MB)
    //   then           : per-half top-k indices (1 MB)
    unsigned short* wsB = (unsigned short*)d_ws;
    size_t wsBElems = (size_t)D_INP * NLAT;                     // bf16 count
    float* wsTopV = (float*)((char*)d_ws + wsBElems * sizeof(unsigned short));
    int*   wsTopI = (int*)(wsTopV + (size_t)BATCH * NSPLIT * TOPK);

    // K1: one-pass W_enc f32->bf16 swizzle into WMMA fragment order.
    dim3 gswz(NLAT / 128, KB);
    sae_swizzle_wenc<<<gswz, THREADS, 0, stream>>>(Wenc, wsB);

    // K2: zero the 402 MB acts region (scatter targets only ~512 KB of it).
    float4* acts4 = (float4*)(out + (size_t)BATCH * D_INP);
    long n4 = (long)BATCH * NLAT / 4;
    sae_zero_acts<<<4096, THREADS, 0, stream>>>(acts4, n4);

    // K3: encoder GEMM (bf16 WMMA) + per-half top-32.  256 WGs.
    sae_encode_topk<<<(BATCH / TILE_M) * NSPLIT, THREADS, 0, stream>>>(
        x, wsB, benc, wsTopV, wsTopI);

    // K4: merge halves -> final top-32, scatter, indices, sparse decode.
    sae_merge_decode<<<BATCH / TILE_M, THREADS, 0, stream>>>(
        wsTopV, wsTopI, Wdec, bdec, out);
}
